// MesaLayer_15075335209572
// MI455X (gfx1250) — matrix operand order repaired and checked by reading, not compiled, not run on hardware
//
#include <hip/hip_runtime.h>
#include <math.h>
#include <stdint.h>

typedef __attribute__((ext_vector_type(16))) __bf16 v16bf;
typedef __attribute__((ext_vector_type(8)))  __bf16 v8bf;
typedef __attribute__((ext_vector_type(8)))  float  v8f;

#define NHEAD   16
#define DKDIM   64
#define DVDIM   64
#define SEQLEN  512
#define NBATCH  2
#define DMODEL  1024
#define CGSTEPS 8
#define LAMREG  1.0f
#define EWEIGHT 0.5f

// ---- CDNA5 async global->LDS copy (ASYNCcnt path), guarded per-toolchain ----
#if defined(__has_builtin)
#if __has_builtin(__builtin_amdgcn_global_load_async_to_lds_b32) && \
    __has_builtin(__builtin_amdgcn_s_wait_asynccnt)
#define USE_ASYNC_LDS 1
#endif
#endif
#ifndef USE_ASYNC_LDS
#define USE_ASYNC_LDS 0
#endif

typedef __attribute__((address_space(1))) int* gint_ptr;
typedef __attribute__((address_space(3))) int* lint_ptr;

__device__ __forceinline__ void async_copy_f32(const float* g, float* l) {
#if USE_ASYNC_LDS
  __builtin_amdgcn_global_load_async_to_lds_b32(
      (gint_ptr)(void*)g, (lint_ptr)(void*)l, 0, 0);
#else
  *l = *g;
#endif
}

__device__ __forceinline__ void async_wait_all() {
#if USE_ASYNC_LDS
  __builtin_amdgcn_s_wait_asynccnt(0);
#endif
}

// ---------------------------------------------------------------------------
// C[M,N] = A[M,K] @ W[K,N]; fp32 in/out, bf16 WMMA math (f32 accumulate).
// Block: 256 threads = 8 wave32. Block tile 128x64, wave tile 32x32
// (2 A-frags x 2 B-frags -> 4 v_wmma per K-step of 32).
// LDS row stride 40 bf16 (80B): 16B-aligned rows, conflict-free bank spread.
// ---------------------------------------------------------------------------
__global__ __launch_bounds__(256)
void wmma_gemm_bf16(const float* __restrict__ A, const float* __restrict__ W,
                    float* __restrict__ C, int M, int N, int K) {
  constexpr int LDT = 40;
  __shared__ __align__(16) __bf16 sA[128 * LDT];      // [m][k]
  __shared__ __align__(16) __bf16 sB[64 * LDT];       // transposed: [n][k]
  const int tid  = threadIdx.x;
  const int wave = tid >> 5;
  const int lane = tid & 31;
  const int half = lane >> 4;
  const int l16  = lane & 15;
  const int wr   = (wave & 3) * 32;   // wave row offset within 128-row tile
  const int wc   = (wave >> 2) * 32;  // wave col offset within 64-col tile
  const int rowBase = blockIdx.y * 128;
  const int colBase = blockIdx.x * 64;

  v8f acc00 = {}, acc01 = {}, acc10 = {}, acc11 = {};

  for (int k0 = 0; k0 < K; k0 += 32) {
    // Prefetch next K-step tiles into cache (lowers to global_prefetch_b8)
    if (k0 + 32 < K) {
      __builtin_prefetch(A + (size_t)(rowBase + (tid >> 1)) * K + k0 + 32 +
                             (tid & 1) * 16, 0, 3);
      __builtin_prefetch(W + (size_t)(k0 + 32 + (tid >> 3)) * N + colBase +
                             (tid & 7) * 8, 0, 3);
    }
    // Stage A tile: 128 rows x 32 k (f32 -> bf16): 1024 float4 / 256 thr
    #pragma unroll
    for (int i = 0; i < 4; ++i) {
      int idx = tid + i * 256;            // 0..1023
      int r   = idx >> 3;                 // 0..127
      int c4  = (idx & 7) << 2;           // 0..28
      float4 f = *(const float4*)(A + (size_t)(rowBase + r) * K + k0 + c4);
      __bf16* d = &sA[r * LDT + c4];
      d[0] = (__bf16)f.x; d[1] = (__bf16)f.y; d[2] = (__bf16)f.z; d[3] = (__bf16)f.w;
    }
    // Stage B tile transposed: 32 k x 64 n -> sB[n][k]
    #pragma unroll
    for (int i = 0; i < 2; ++i) {
      int idx = tid + i * 256;
      int r   = idx >> 4;                 // k: 0..31
      int c4  = (idx & 15) << 2;          // n: 0..60
      float4 f = *(const float4*)(W + (size_t)(k0 + r) * N + colBase + c4);
      sB[(c4 + 0) * LDT + r] = (__bf16)f.x;
      sB[(c4 + 1) * LDT + r] = (__bf16)f.y;
      sB[(c4 + 2) * LDT + r] = (__bf16)f.z;
      sB[(c4 + 3) * LDT + r] = (__bf16)f.w;
    }
    __syncthreads();

    // A fragments (16x32 bf16 layout): lanes 0-15 -> K {0..7,16..23},
    //                                  lanes 16-31 -> K {8..15,24..31}
    union { v16bf v; v8bf h[2]; } a0, a1, b0, b1;
    {
      const __bf16* ar0 = &sA[(wr + l16) * LDT];
      a0.h[0] = *(const v8bf*)(ar0 + half * 8);
      a0.h[1] = *(const v8bf*)(ar0 + 16 + half * 8);
      const __bf16* ar1 = &sA[(wr + 16 + l16) * LDT];
      a1.h[0] = *(const v8bf*)(ar1 + half * 8);
      a1.h[1] = *(const v8bf*)(ar1 + 16 + half * 8);
    }
    // B fragments (32x16): lane n=l16; elements = K {half*8..+7, 16+half*8..+7}
    {
      const __bf16* br0 = &sB[(wc + l16) * LDT];
      b0.h[0] = *(const v8bf*)(br0 + half * 8);
      b0.h[1] = *(const v8bf*)(br0 + 16 + half * 8);
      const __bf16* br1 = &sB[(wc + 16 + l16) * LDT];
      b1.h[0] = *(const v8bf*)(br1 + half * 8);
      b1.h[1] = *(const v8bf*)(br1 + 16 + half * 8);
    }
    acc00 = __builtin_amdgcn_wmma_f32_16x16x32_bf16(false, a0.v, false, b0.v,
                                                    (short)0, acc00, false, false);
    acc01 = __builtin_amdgcn_wmma_f32_16x16x32_bf16(false, a0.v, false, b1.v,
                                                    (short)0, acc01, false, false);
    acc10 = __builtin_amdgcn_wmma_f32_16x16x32_bf16(false, a1.v, false, b0.v,
                                                    (short)0, acc10, false, false);
    acc11 = __builtin_amdgcn_wmma_f32_16x16x32_bf16(false, a1.v, false, b1.v,
                                                    (short)0, acc11, false, false);
    __syncthreads();
  }

  // D layout: VGPR r -> row (+ half*8 + r); col = lane&15
  #pragma unroll
  for (int r = 0; r < 8; ++r) {
    int row0 = rowBase + wr + half * 8 + r;
    int row1 = row0 + 16;
    C[(size_t)row0 * N + colBase + wc + l16]      = acc00[r];
    C[(size_t)row0 * N + colBase + wc + 16 + l16] = acc01[r];
    C[(size_t)row1 * N + colBase + wc + l16]      = acc10[r];
    C[(size_t)row1 * N + colBase + wc + 16 + l16] = acc11[r];
  }
}

// ---------------------------------------------------------------------------
// Gates: gamma = sigmoid(x@w_gamma); beta = min(sigmoid(x@w_beta), gamma);
// beta = min(beta * (1 + EW*sigmoid(efas*w_energy)), gamma)
// ---------------------------------------------------------------------------
__global__ __launch_bounds__(256)
void gates_kernel(const float* __restrict__ x, const float* __restrict__ efas,
                  const float* __restrict__ w_gamma, const float* __restrict__ w_beta,
                  const float* __restrict__ w_energy,
                  float* __restrict__ gamma, float* __restrict__ beta) {
  int gidx = blockIdx.x * 256 + threadIdx.x;    // row*NHEAD + h
  if (gidx >= NBATCH * SEQLEN * NHEAD) return;
  int row = gidx >> 4;
  int h   = gidx & 15;
  const float* xr = x + (size_t)row * DMODEL;
  float sg = 0.f, sb = 0.f;
  for (int i = 0; i < DMODEL; ++i) {
    float xv = xr[i];
    sg += xv * w_gamma[i * NHEAD + h];
    sb += xv * w_beta[i * NHEAD + h];
  }
  float g = 1.f / (1.f + __expf(-sg));
  float b = 1.f / (1.f + __expf(-sb));
  b = fminf(b, g);
  float eg = 1.f / (1.f + __expf(-efas[row] * w_energy[h]));
  b = fminf(b * (1.f + EWEIGHT * eg), g);
  gamma[gidx] = g;
  beta[gidx]  = b;
}

// ---------------------------------------------------------------------------
// Fused recurrence + CG + output matvec. One block per (b,h); H/G stay in LDS
// for the whole sequence. q/k/v/gates for step s+1 are double-buffered into
// LDS with async global->LDS copies while step s's CG solve runs.
// 256 threads: thread t owns row i=t>>2, 16-wide j slice (t&3)*16.
// ---------------------------------------------------------------------------
// stage layout: [0..63]=q  [64..127]=k  [128..191]=v  [192]=gamma  [193]=beta
#define STG 200

__device__ __forceinline__
void prefetch_step(const float* Q, const float* Km, const float* V,
                   const float* gamma, const float* beta,
                   size_t base, int gidx, float* st, int tid) {
  if (tid < 64) {
    async_copy_f32(Q  + base + tid, st + tid);
    async_copy_f32(Km + base + tid, st + 64 + tid);
    async_copy_f32(V  + base + tid, st + 128 + tid);
  } else if (tid == 64) {
    async_copy_f32(gamma + gidx, st + 192);
    async_copy_f32(beta  + gidx, st + 193);
  }
}

__global__ __launch_bounds__(256)
void mesa_scan_cg(const float* __restrict__ Q, const float* __restrict__ Km,
                  const float* __restrict__ V, const float* __restrict__ gamma,
                  const float* __restrict__ beta, float* __restrict__ O) {
  const int bh = blockIdx.x;          // 0..31
  const int b  = bh >> 4;
  const int h  = bh & 15;
  const int tid = threadIdx.x;

  __shared__ float H[DKDIM * 65];     // pad to 65 to de-conflict banks
  __shared__ float G[DVDIM * 65];
  __shared__ float stage[2][STG];
  __shared__ float sol[DKDIM], rr[DKDIM], pp[DKDIM], Ap[DKDIM];
  __shared__ float part[DKDIM * 4];
  __shared__ float scal[4];           // [0]=rsold  [1]=alpha  [2]=betaCG

  for (int idx = tid; idx < DKDIM * 65; idx += 256) { H[idx] = 0.f; G[idx] = 0.f; }

  const int i  = tid >> 2;
  const int j0 = (tid & 3) * 16;

  // kick off prefetch of s=0
  {
    const size_t base0 = ((size_t)(b * SEQLEN)) * (NHEAD * DKDIM) + (size_t)h * DKDIM;
    prefetch_step(Q, Km, V, gamma, beta, base0, (b * SEQLEN) * NHEAD + h,
                  stage[0], tid);
  }
  __syncthreads();

  for (int s = 0; s < SEQLEN; ++s) {
    float* st = stage[s & 1];
    const size_t base = ((size_t)(b * SEQLEN + s)) * (NHEAD * DKDIM) + (size_t)h * DKDIM;

    async_wait_all();       // stage[s&1] resident
    __syncthreads();

    // overlap: issue async copies for step s+1 into the other buffer
    if (s + 1 < SEQLEN) {
      const size_t nbase = base + NHEAD * DKDIM;
      prefetch_step(Q, Km, V, gamma, beta, nbase, (b * SEQLEN + s + 1) * NHEAD + h,
                    stage[(s + 1) & 1], tid);
    }

    const float* qv = st;
    const float* kv = st + 64;
    const float* vv = st + 128;
    const float g  = st[192];
    const float be = st[193];

    if (tid < DKDIM) {
      float q = qv[tid];
      rr[tid]  = q;
      pp[tid]  = q;
      sol[tid] = 0.f;
    }

    // Rank-1 decay update: H = g*H + be*k k^T ; G = g*G + be*v k^T
    {
      float ki = kv[i];
      float vi = vv[i];
      #pragma unroll
      for (int e = 0; e < 16; ++e) {
        int j = j0 + e;
        float kj = kv[j];
        H[i * 65 + j] = g * H[i * 65 + j] + be * ki * kj;
        G[i * 65 + j] = g * G[i * 65 + j] + be * vi * kj;
      }
    }
    __syncthreads();

    // rsold = <q,q>  (wave 0, wave32 shuffle reduction)
    if (tid < 32) {
      float v2 = qv[tid] * qv[tid] + qv[tid + 32] * qv[tid + 32];
      #pragma unroll
      for (int m = 16; m > 0; m >>= 1) v2 += __shfl_xor(v2, m, 32);
      if (tid == 0) scal[0] = v2;
    }
    __syncthreads();

    for (int it = 0; it < CGSTEPS; ++it) {
      // Ap = (H + lam I) p : 4 partial sums per row, then combine
      float partial = 0.f;
      #pragma unroll
      for (int e = 0; e < 16; ++e) partial += H[i * 65 + j0 + e] * pp[j0 + e];
      part[i * 4 + (tid & 3)] = partial;
      __syncthreads();
      if (tid < DKDIM)
        Ap[tid] = part[tid * 4] + part[tid * 4 + 1] + part[tid * 4 + 2] +
                  part[tid * 4 + 3] + LAMREG * pp[tid];
      __syncthreads();
      // alpha = rsold / (<p,Ap> + eps)
      if (tid < 32) {
        float pa = pp[tid] * Ap[tid] + pp[tid + 32] * Ap[tid + 32];
        #pragma unroll
        for (int m = 16; m > 0; m >>= 1) pa += __shfl_xor(pa, m, 32);
        if (tid == 0) scal[1] = scal[0] / (pa + 1e-10f);
      }
      __syncthreads();
      const float alpha = scal[1];
      if (tid < DKDIM) {
        sol[tid] += alpha * pp[tid];
        rr[tid]  -= alpha * Ap[tid];
      }
      __syncthreads();
      // betaCG = rsnew / (rsold + eps); rsold = rsnew
      if (tid < 32) {
        float rn = rr[tid] * rr[tid] + rr[tid + 32] * rr[tid + 32];
        #pragma unroll
        for (int m = 16; m > 0; m >>= 1) rn += __shfl_xor(rn, m, 32);
        if (tid == 0) { scal[2] = rn / (scal[0] + 1e-10f); scal[0] = rn; }
      }
      __syncthreads();
      if (tid < DKDIM) pp[tid] = rr[tid] + scal[2] * pp[tid];
      __syncthreads();
    }

    // o = G @ sol
    {
      float partial = 0.f;
      #pragma unroll
      for (int e = 0; e < 16; ++e) partial += G[i * 65 + j0 + e] * sol[j0 + e];
      part[i * 4 + (tid & 3)] = partial;
    }
    __syncthreads();
    if (tid < DKDIM)
      O[base + tid] = part[tid * 4] + part[tid * 4 + 1] +
                      part[tid * 4 + 2] + part[tid * 4 + 3];
    __syncthreads();
  }
}

// ---------------------------------------------------------------------------
extern "C" void kernel_launch(void* const* d_in, const int* in_sizes, int n_in,
                              void* d_out, int out_size, void* d_ws, size_t ws_size,
                              hipStream_t stream) {
  (void)in_sizes; (void)n_in; (void)out_size; (void)ws_size;
  const float* x        = (const float*)d_in[0];
  const float* efas     = (const float*)d_in[1];
  const float* w_q      = (const float*)d_in[2];
  const float* w_k      = (const float*)d_in[3];
  const float* w_v      = (const float*)d_in[4];
  const float* w_gamma  = (const float*)d_in[5];
  const float* w_beta   = (const float*)d_in[6];
  const float* w_energy = (const float*)d_in[7];
  const float* w_o      = (const float*)d_in[8];
  float* out = (float*)d_out;

  const int M  = NBATCH * SEQLEN;   // 1024
  const int N  = NHEAD * DKDIM;     // 1024
  const int Kd = DMODEL;            // 1024

  float* ws = (float*)d_ws;
  float* Qb = ws;
  float* Kb = Qb + (size_t)M * N;
  float* Vb = Kb + (size_t)M * N;
  float* Ob = Vb + (size_t)M * N;
  float* Gm = Ob + (size_t)M * N;
  float* Bt = Gm + (size_t)M * NHEAD;

  dim3 gemmGrid(N / 64, M / 128);
  wmma_gemm_bf16<<<gemmGrid, 256, 0, stream>>>(x, w_q, Qb, M, N, Kd);
  wmma_gemm_bf16<<<gemmGrid, 256, 0, stream>>>(x, w_k, Kb, M, N, Kd);
  wmma_gemm_bf16<<<gemmGrid, 256, 0, stream>>>(x, w_v, Vb, M, N, Kd);
  gates_kernel<<<(M * NHEAD + 255) / 256, 256, 0, stream>>>(x, efas, w_gamma,
                                                            w_beta, w_energy, Gm, Bt);
  mesa_scan_cg<<<NBATCH * NHEAD, 256, 0, stream>>>(Qb, Kb, Vb, Gm, Bt, Ob);
  wmma_gemm_bf16<<<dim3(DMODEL / 64, M / 128), 256, 0, stream>>>(Ob, w_o, out,
                                                                 M, DMODEL, N);
}
